// MultiHeadAttention_64364379898327
// MI455X (gfx1250) — compile-verified, matrix operand order checked by reading, not run on hardware
//
#include <hip/hip_runtime.h>

#define B_  2
#define L_  2048
#define D_  1024
#define H_  16
#define DH_ 64

typedef __attribute__((ext_vector_type(16))) __bf16 v16bf;
typedef __attribute__((ext_vector_type(8)))  __bf16 v8bf;
typedef __attribute__((ext_vector_type(8)))  float  v8f;

// ---------- helpers ----------

static __device__ __forceinline__ unsigned short f2bf(float f) {
  unsigned int u = __builtin_bit_cast(unsigned int, f);
  u += 0x7FFFu + ((u >> 16) & 1u);           // round-to-nearest-even
  return (unsigned short)(u >> 16);
}

// Low 32 bits of a generic pointer to LDS == wave-relative LDS byte address
// (ISA 10.2: LDS aperture).
static __device__ __forceinline__ unsigned int ldsoff(const void* p) {
  return (unsigned int)(unsigned long long)(uintptr_t)p;
}

// GLOBAL_LOAD_ASYNC_TO_LDS_B128: 16B per lane, memory -> LDS, ASYNCcnt-tracked.
static __device__ __forceinline__ void async_load_b128(unsigned int lds_addr,
                                                       const void* gptr) {
  asm volatile("global_load_async_to_lds_b128 %0, %1, off"
               :: "v"(lds_addr), "v"((unsigned long long)(uintptr_t)gptr)
               : "memory");
}

template <int N>
static __device__ __forceinline__ void wait_asynccnt() {
  asm volatile("s_wait_asynccnt %0" :: "n"(N) : "memory");
}

// Build a 16x32 bf16 A/B fragment for v_wmma_f32_16x16x32_bf16.
// Per ISA 7.12.2: lane = row + 16*khalf; elements 0..7 hold K = 8*khalf+0..7,
// elements 8..15 hold K = 16+8*khalf+0..7  -> two 16B reads per lane.
static __device__ __forceinline__ v16bf load_frag(const unsigned short* p0,
                                                  const unsigned short* p1) {
  v8bf lo = *(const v8bf*)(const void*)p0;
  v8bf hi = *(const v8bf*)(const void*)p1;
  v16bf r;
#pragma unroll
  for (int i = 0; i < 8; ++i) { r[i] = lo[i]; r[i + 8] = hi[i]; }
  return r;
}

static __device__ __forceinline__ v8f wmma_bf16(v16bf a, v16bf b, v8f c) {
  return __builtin_amdgcn_wmma_f32_16x16x32_bf16(false, a, false, b,
                                                 (short)0, c, false, false);
}

// ---------- f32 -> bf16 conversion ----------

__global__ void cvt_f32_bf16(const float* __restrict__ in,
                             unsigned short* __restrict__ out, int n) {
  int i = (blockIdx.x * 256 + threadIdx.x) * 4;
  if (i + 3 < n) {
    float4 f = *(const float4*)(in + i);
    ushort4 r;
    r.x = f2bf(f.x); r.y = f2bf(f.y); r.z = f2bf(f.z); r.w = f2bf(f.w);
    *(ushort4*)(out + i) = r;
  }
}

// ---------- NT GEMM: C[M,N] = A[M,K] * W[N,K]^T, bf16 in, f32 acc ----------
// Double-buffered LDS tiles filled with global_load_async_to_lds_b128.
// mode 0: write bf16, head-split Q/K layout [bh][L][DH]
// mode 2: write bf16, transposed V layout  [bh][DH][L]
// mode 3: write f32 row-major [M, D_]
__global__ void __launch_bounds__(256)
gemm_nt_bf16(const unsigned short* __restrict__ A,
             const unsigned short* __restrict__ W,
             unsigned short* __restrict__ out_bf,
             float* __restrict__ out_f32,
             int K, int mode) {
  __shared__ unsigned short lA[2][128 * 32];
  __shared__ unsigned short lB[2][128 * 32];
  const int tid = threadIdx.x;
  const int lane = tid & 31, w = tid >> 5;
  const int half = lane >> 4, row16 = lane & 15;
  const int m0 = blockIdx.x * 128, n0 = blockIdx.y * 128;
  const int wr = w >> 2, wc = w & 3;      // 2x4 wave grid: 64 rows x 32 cols each

  // per-thread copy slots: rows cr and cr+64, one 16B chunk each for A and B
  const int cr = tid >> 2;                 // 0..63
  const int cc = (tid & 3) * 8;            // half offset within 32-wide K slab
  const unsigned short* gA = A + (size_t)(m0 + cr) * K + cc;
  const unsigned short* gW = W + (size_t)(n0 + cr) * K + cc;
  const size_t rstep = (size_t)64 * K;     // +64 rows in halfs
  const unsigned int la0 = ldsoff(&lA[0][cr * 32 + cc]);
  const unsigned int lb0 = ldsoff(&lB[0][cr * 32 + cc]);
  const unsigned int rowb = 64 * 32 * 2;   // +64 rows in LDS bytes
  const unsigned int bufb = 128 * 32 * 2;  // bytes per LDS buffer

  // prologue: async-load tile 0 into buffer 0
  async_load_b128(la0,        gA);
  async_load_b128(la0 + rowb, gA + rstep);
  async_load_b128(lb0,        gW);
  async_load_b128(lb0 + rowb, gW + rstep);

  v8f acc[4][2];
#pragma unroll
  for (int i = 0; i < 4; ++i)
#pragma unroll
    for (int j = 0; j < 2; ++j) acc[i][j] = (v8f){0,0,0,0,0,0,0,0};

  const int steps = K / 32;
  for (int s = 0; s < steps; ++s) {
    const int cur = s & 1;
    __syncthreads();                       // prev readers of buf[1-cur] done
    if (s + 1 < steps) {
      const unsigned int nb = ((s + 1) & 1) * bufb;
      const unsigned short* ga = gA + (size_t)(s + 1) * 32;
      const unsigned short* gw = gW + (size_t)(s + 1) * 32;
      async_load_b128(la0 + nb,        ga);
      async_load_b128(la0 + nb + rowb, ga + rstep);
      async_load_b128(lb0 + nb,        gw);
      async_load_b128(lb0 + nb + rowb, gw + rstep);
      wait_asynccnt<4>();                  // tile s (this wave's 4 ops) complete
    } else {
      wait_asynccnt<0>();
    }
    __syncthreads();                       // all waves' tile s visible

    const unsigned short* lAc = &lA[cur][0];
    const unsigned short* lBc = &lB[cur][0];
    v16bf bfrag[2];
#pragma unroll
    for (int j = 0; j < 2; ++j) {
      const unsigned short* bp = &lBc[(wc * 32 + j * 16 + row16) * 32];
      bfrag[j] = load_frag(bp + 8 * half, bp + 16 + 8 * half);
    }
#pragma unroll
    for (int i = 0; i < 4; ++i) {
      const unsigned short* ap = &lAc[(wr * 64 + i * 16 + row16) * 32];
      v16bf af = load_frag(ap + 8 * half, ap + 16 + 8 * half);
#pragma unroll
      for (int j = 0; j < 2; ++j) acc[i][j] = wmma_bf16(af, bfrag[j], acc[i][j]);
    }
  }

#pragma unroll
  for (int i = 0; i < 4; ++i)
#pragma unroll
    for (int j = 0; j < 2; ++j)
#pragma unroll
      for (int r = 0; r < 8; ++r) {
        int m = m0 + wr * 64 + i * 16 + r + 8 * half;   // C row (ISA C/D layout)
        int c = n0 + wc * 32 + j * 16 + row16;          // C col
        float v = acc[i][j][r];
        if (mode == 3) {
          out_f32[(size_t)m * D_ + c] = v;
        } else {
          int b = m >> 11, l = m & (L_ - 1);
          int h = c >> 6,  dh = c & 63;
          size_t idx;
          if (mode == 2) idx = (((size_t)(b * H_ + h)) * DH_ + dh) * L_ + l; // V^T
          else           idx = (((size_t)(b * H_ + h)) * L_ + l) * DH_ + dh; // Q/K
          out_bf[idx] = f2bf(v);
        }
      }
}

// ---------- causal flash attention ----------
// Q,K: [bh][L][DH] bf16; Vt: [bh][DH][L] bf16; Ob: [B*L, D] bf16
__global__ void __launch_bounds__(256)
flash_attn(const unsigned short* __restrict__ Q,
           const unsigned short* __restrict__ Kg,
           const unsigned short* __restrict__ Vt,
           unsigned short* __restrict__ Ob) {
  __shared__ unsigned short kT[2][32 * 64];   // 32 keys x 64 dh
  __shared__ unsigned short vT[2][64 * 32];   // 64 dh   x 32 keys (V^T rows)
  __shared__ unsigned short pT[8][16 * 32];   // wave-private P tiles

  const int tid = threadIdx.x, lane = tid & 31, w = tid >> 5;
  const int half = lane >> 4, row16 = lane & 15;
  const int bh = blockIdx.y, b = bh >> 4, h = bh & 15;
  const int q0 = blockIdx.x * 128;
  const int qr = q0 + w * 16;                 // this wave's 16 query rows

  const unsigned short* qh = Q  + (size_t)bh * L_ * DH_;
  const unsigned short* kh = Kg + (size_t)bh * L_ * DH_;
  const unsigned short* vh = Vt + (size_t)bh * DH_ * L_;

  // per-thread async-copy slots for K and V^T blocks
  const int kr = tid >> 3, kc = (tid & 7) * 8;       // kT: 32 rows x 8 chunks
  const int vr = tid >> 2, vc = (tid & 3) * 8;       // vT: 64 rows x 4 chunks
  const unsigned short* gk = kh + (size_t)kr * DH_ + kc;  // + kb*32*DH_ per block
  const unsigned short* gv = vh + (size_t)vr * L_ + vc;   // + kb*32     per block
  const unsigned int lk0 = ldsoff(&kT[0][kr * 64 + kc]);
  const unsigned int lv0 = ldsoff(&vT[0][vr * 32 + vc]);
  const unsigned int kvb = 32 * 64 * 2;               // 4KB per buffer (both)

  // preload Q fragments (dh 0..31 and 32..63) straight from global
  const unsigned short* qrow = qh + (size_t)(qr + row16) * DH_;
  v16bf qf0 = load_frag(qrow + 8 * half,       qrow + 16 + 8 * half);
  v16bf qf1 = load_frag(qrow + 32 + 8 * half,  qrow + 48 + 8 * half);

  // prologue: async-load key-block 0 into buffer 0
  async_load_b128(lk0, gk);
  async_load_b128(lv0, gv);

  v8f o[4];
#pragma unroll
  for (int d = 0; d < 4; ++d) o[d] = (v8f){0,0,0,0,0,0,0,0};
  float mrun[8], lrun[8];
#pragma unroll
  for (int r = 0; r < 8; ++r) { mrun[r] = -3.0e38f; lrun[r] = 0.0f; }

  const int nkb = (blockIdx.x + 1) * 4;       // causal: keys only up to q-tile end
  for (int kb = 0; kb < nkb; ++kb) {
    const int k0 = kb * 32;
    const int cur = kb & 1;
    __syncthreads();                          // prev readers of buf[1-cur] done
    if (kb + 1 < nkb) {
      const unsigned int nb = ((kb + 1) & 1) * kvb;
      async_load_b128(lk0 + nb, gk + (size_t)(kb + 1) * 32 * DH_);
      async_load_b128(lv0 + nb, gv + (size_t)(kb + 1) * 32);
      wait_asynccnt<2>();                     // block kb complete
    } else {
      wait_asynccnt<0>();
    }
    __syncthreads();                          // all waves' block kb visible

    // S = Q * K^T  (two 16x16 tiles, dh contraction split 32+32)
    v8f s[2];
#pragma unroll
    for (int j = 0; j < 2; ++j) {
      const unsigned short* kp = &kT[cur][(j * 16 + row16) * 64];
      v16bf b0 = load_frag(kp + 8 * half,      kp + 16 + 8 * half);
      v16bf b1 = load_frag(kp + 32 + 8 * half, kp + 48 + 8 * half);
      v8f z = (v8f){0,0,0,0,0,0,0,0};
      z = wmma_bf16(qf0, b0, z);
      z = wmma_bf16(qf1, b1, z);
      s[j] = z;
    }

    // scale, causal mask, online softmax (row = r + 8*half, col = row16)
    float pv[2][8], corr[8];
#pragma unroll
    for (int r = 0; r < 8; ++r) {
      int qi = qr + r + 8 * half;
      float rm = -3.0e38f;
#pragma unroll
      for (int j = 0; j < 2; ++j) {
        int ki = k0 + j * 16 + row16;
        float v = s[j][r] * 0.125f;            // 1/sqrt(64)
        if (ki > qi) v = -3.0e38f;
        s[j][r] = v;
        rm = fmaxf(rm, v);
      }
#pragma unroll
      for (int off = 1; off < 16; off <<= 1)   // reduce over 16 lanes of half
        rm = fmaxf(rm, __shfl_xor(rm, off, 32));
      float mn = fmaxf(mrun[r], rm);
      float cr2 = __expf(mrun[r] - mn);
      mrun[r] = mn;
      corr[r] = cr2;
      float rs = 0.0f;
#pragma unroll
      for (int j = 0; j < 2; ++j) {
        float p = __expf(s[j][r] - mn);
        pv[j][r] = p;
        rs += p;
      }
#pragma unroll
      for (int off = 1; off < 16; off <<= 1)
        rs += __shfl_xor(rs, off, 32);
      lrun[r] = lrun[r] * cr2 + rs;
    }
#pragma unroll
    for (int d = 0; d < 4; ++d)
#pragma unroll
      for (int r = 0; r < 8; ++r) o[d][r] *= corr[r];

    // re-layout P (C/D layout) -> A-fragment layout through wave-private LDS
#pragma unroll
    for (int j = 0; j < 2; ++j)
#pragma unroll
      for (int r = 0; r < 8; ++r)
        pT[w][(r + 8 * half) * 32 + j * 16 + row16] = f2bf(pv[j][r]);
    asm volatile("s_wait_dscnt 0" ::: "memory");
    const unsigned short* pp = &pT[w][row16 * 32];
    v16bf pf = load_frag(pp + 8 * half, pp + 16 + 8 * half);

    // O += P * V   (NT against V^T rows)
#pragma unroll
    for (int d = 0; d < 4; ++d) {
      const unsigned short* vp = &vT[cur][(d * 16 + row16) * 32];
      v16bf bf = load_frag(vp + 8 * half, vp + 16 + 8 * half);
      o[d] = wmma_bf16(pf, bf, o[d]);
    }
  }

  // normalize and store to [B*L, D] bf16 (feeds the Wo projection)
#pragma unroll
  for (int r = 0; r < 8; ++r) {
    float inv = 1.0f / lrun[r];
    int l = qr + r + 8 * half;
    size_t rowbase = ((size_t)b * L_ + l) * D_;
#pragma unroll
    for (int d = 0; d < 4; ++d) {
      int c = h * DH_ + d * 16 + row16;
      Ob[rowbase + c] = f2bf(o[d][r] * inv);
    }
  }
}

// ---------- launcher ----------

extern "C" void kernel_launch(void* const* d_in, const int* in_sizes, int n_in,
                              void* d_out, int out_size, void* d_ws, size_t ws_size,
                              hipStream_t stream) {
  (void)in_sizes; (void)n_in; (void)out_size; (void)ws_size;
  const float* x  = (const float*)d_in[0];
  // d_in[1] = mask (tril) -> causality applied analytically in flash_attn
  const float* Wq = (const float*)d_in[2];
  const float* Wk = (const float*)d_in[3];
  const float* Wv = (const float*)d_in[4];
  const float* Wo = (const float*)d_in[5];
  float* out = (float*)d_out;

  const size_t nx = (size_t)B_ * L_ * D_;   // 4096*1024
  const size_t nw = (size_t)D_ * D_;        // 1024*1024

  char* ws = (char*)d_ws;
  unsigned short* xb  = (unsigned short*)ws; ws += nx * 2;
  unsigned short* wqb = (unsigned short*)ws; ws += nw * 2;
  unsigned short* wkb = (unsigned short*)ws; ws += nw * 2;
  unsigned short* wvb = (unsigned short*)ws; ws += nw * 2;
  unsigned short* wob = (unsigned short*)ws; ws += nw * 2;
  unsigned short* qb  = (unsigned short*)ws; ws += nx * 2;
  unsigned short* kbf = (unsigned short*)ws; ws += nx * 2;
  unsigned short* vtb = (unsigned short*)ws; ws += nx * 2;
  unsigned short* ab  = xb;                  // x no longer needed after projections

  cvt_f32_bf16<<<(unsigned)(nx / 1024), 256, 0, stream>>>(x,  xb,  (int)nx);
  cvt_f32_bf16<<<(unsigned)(nw / 1024), 256, 0, stream>>>(Wq, wqb, (int)nw);
  cvt_f32_bf16<<<(unsigned)(nw / 1024), 256, 0, stream>>>(Wk, wkb, (int)nw);
  cvt_f32_bf16<<<(unsigned)(nw / 1024), 256, 0, stream>>>(Wv, wvb, (int)nw);
  cvt_f32_bf16<<<(unsigned)(nw / 1024), 256, 0, stream>>>(Wo, wob, (int)nw);

  dim3 gg(32, 8);   // M/128 x N/128
  gemm_nt_bf16<<<gg, 256, 0, stream>>>(xb, wqb, qb,  nullptr, D_, 0);  // Q
  gemm_nt_bf16<<<gg, 256, 0, stream>>>(xb, wkb, kbf, nullptr, D_, 0);  // K
  gemm_nt_bf16<<<gg, 256, 0, stream>>>(xb, wvb, vtb, nullptr, D_, 2);  // V^T

  dim3 ga(L_ / 128, B_ * H_);
  flash_attn<<<ga, 256, 0, stream>>>(qb, kbf, vtb, ab);

  gemm_nt_bf16<<<gg, 256, 0, stream>>>(ab, wob, nullptr, out, D_, 3);  // out proj
}
